// MultiheadSelfAttention_90915867721901
// MI455X (gfx1250) — matrix operand order repaired and checked by reading, not compiled, not run on hardware
//
#include <hip/hip_runtime.h>

typedef __attribute__((ext_vector_type(16))) __bf16 v16bf;
typedef __attribute__((ext_vector_type(8)))  __bf16 v8bf;
typedef __attribute__((ext_vector_type(4)))  __bf16 v4bf;
typedef __attribute__((ext_vector_type(8)))  float  v8f;
typedef __attribute__((ext_vector_type(4)))  float  v4f;

typedef __attribute__((ext_vector_type(4))) unsigned int u32x4;
typedef __attribute__((ext_vector_type(8))) int          i32x8;
typedef __attribute__((ext_vector_type(4))) int          i32x4;

#if defined(__gfx1250__) && __has_builtin(__builtin_amdgcn_tensor_load_to_lds) && \
    __has_builtin(__builtin_amdgcn_s_wait_tensorcnt)
#define USE_TDM 1
#else
#define USE_TDM 0
#endif

namespace {
constexpr int S_LEN  = 2048;
constexpr int DMODEL = 1024;
constexpr int NH     = 16;
constexpr int DH     = 64;
constexpr int BATCH  = 4;
constexpr int ROWS   = BATCH * S_LEN;   // 8192
}

// ---------------------------------------------------------------- fp32->bf16
__global__ void k_cvt_bf16(const float* __restrict__ src, __bf16* __restrict__ dst, int n)
{
    int i = (blockIdx.x * blockDim.x + threadIdx.x) * 4;
    if (i >= n) return;
    v4f v = *(const v4f*)(src + i);
    v4bf o;
    o[0] = (__bf16)v.x; o[1] = (__bf16)v.y; o[2] = (__bf16)v.z; o[3] = (__bf16)v.w;
    *(v4bf*)(dst + i) = o;
}

// ---------------------------------------------------------------- TDM tile load
// DMA a 128-row x 32-col bf16 tile (row stride = kdim elements) into LDS,
// padding 8 DWORDs (16 bf16) after every 16 DWORDs (32 bf16 = one tile row)
// so the LDS image has row stride LDT = 48 bf16.  D# layout per CDNA5 ISA
// CDNA5 ISA.  Issue from one wave; EXEC is ignored.
__device__ __forceinline__ void tdm_load_tile_128x32(const __bf16* gtile,
                                                     unsigned ldsOff,
                                                     int kdim, int rowsTot)
{
#if USE_TDM
    unsigned long long ga = (unsigned long long)(uintptr_t)gtile;
    u32x4 g0;
    g0[0] = 1u;                                                 // count = 1 valid D#
    g0[1] = ldsOff;                                             // LDS byte address
    g0[2] = (unsigned)(ga & 0xFFFFFFFFu);                       // global_addr[31:0]
    g0[3] = (unsigned)((ga >> 32) & 0x01FFFFFFu) | (2u << 30);  // [56:32] | type=2
    i32x8 g1;
    g1[0] = (1 << 16)        // data_size = 1 -> 2 bytes
          | (1 << 20)        // pad_enable
          | (3 << 22)        // pad_interval: code 3 -> every 16 DWORDs
          | (7 << 25);       // pad_amount:   code 7 -> 8 DWORDs
    g1[1] = (kdim & 0xFFFF) << 16;                              // tensor_dim0 lo16
    g1[2] = ((kdim >> 16) & 0xFFFF) | ((rowsTot & 0xFFFF) << 16); // dim0 hi | dim1 lo
    g1[3] = ((rowsTot >> 16) & 0xFFFF) | (32 << 16);            // dim1 hi | tile_dim0=32
    g1[4] = 128;                                                // tile_dim1=128, tile_dim2=0
    g1[5] = kdim;                                               // tensor_dim0_stride lo32
    g1[6] = 0;                                                  // stride hi | dim1_stride lo
    g1[7] = 0;
    i32x4 gz4 = {0, 0, 0, 0};
    i32x8 gz8 = {0, 0, 0, 0, 0, 0, 0, 0};
    __builtin_amdgcn_tensor_load_to_lds(g0, g1, gz4, gz4, gz8, 0);
#else
    (void)gtile; (void)ldsOff; (void)kdim; (void)rowsTot;
#endif
}

// ---------------------------------------------------------------- GEMM
// C[M,N] = A[M,K] * B[N,K]^T   (einsum 'bsd,ed->bse': B rows are output cols)
// 128x128 tile, double-buffered KT=32 slabs fed by the TDM: while the 8 waves
// run WMMAs on buffer i, wave 0 has the DMA for buffer i^1 in flight; with
// in-order TENSORcnt, s_wait_tensorcnt(2) retires exactly the current tiles.
// MODE 0: store bf16 scattered to [b, h, s, dh]   (QKV projections)
// MODE 1: store fp32 row-major [row, col]         (final wo projection)
template <int MODE>
__global__ __launch_bounds__(256) void k_gemm(const __bf16* __restrict__ A,
                                              const __bf16* __restrict__ B,
                                              void* __restrict__ C,
                                              int Kdim, int Ndim, int Mdim)
{
    constexpr int BT = 128, KT = 32, LDT = KT + 16;     // LDT = 48 bf16 = 96B stride
    __shared__ alignas(32) __bf16 lA[2][BT * LDT];      // 2 x 12KB
    __shared__ alignas(32) __bf16 lB[2][BT * LDT];      // 2 x 12KB   (48KB total)

    const int tid  = threadIdx.x;
    const int lane = tid & 31, wave = tid >> 5;
    const int wm = wave & 1, wn = wave >> 1;            // 2 x 4 wave grid
    const int hl = lane >> 4, l16 = lane & 15;
    const int tM = blockIdx.y * BT, tN = blockIdx.x * BT;

    const v8f vzero = {0.f, 0.f, 0.f, 0.f, 0.f, 0.f, 0.f, 0.f};
    v8f acc[4][2];
#pragma unroll
    for (int i = 0; i < 4; ++i)
#pragma unroll
        for (int j = 0; j < 2; ++j) acc[i][j] = vzero;

#if USE_TDM
    const unsigned laOff0 = (unsigned)(uintptr_t)(void*)&lA[0][0];
    const unsigned laOff1 = (unsigned)(uintptr_t)(void*)&lA[1][0];
    const unsigned lbOff0 = (unsigned)(uintptr_t)(void*)&lB[0][0];
    const unsigned lbOff1 = (unsigned)(uintptr_t)(void*)&lB[1][0];
    if (wave == 0) {                                    // prologue: DMA slab 0
        tdm_load_tile_128x32(A + (size_t)tM * Kdim, laOff0, Kdim, Mdim);
        tdm_load_tile_128x32(B + (size_t)tN * Kdim, lbOff0, Kdim, Ndim);
    }
#else
    const int cpRow = tid >> 1;            // 0..127: one LDS row per 2 threads
    const int cpOff = (tid & 1) * 16;      // 16 bf16 (32B) per thread per tile
#endif

    const int nk = Kdim / KT;
    for (int i = 0; i < nk; ++i) {
        const int buf = i & 1;
#if USE_TDM
        if (wave == 0) {
            if (i + 1 < nk) {              // DMA next slab into the other buffer
                tdm_load_tile_128x32(A + (size_t)tM * Kdim + (i + 1) * KT,
                                     buf ? laOff0 : laOff1, Kdim, Mdim);
                tdm_load_tile_128x32(B + (size_t)tN * Kdim + (i + 1) * KT,
                                     buf ? lbOff0 : lbOff1, Kdim, Ndim);
                __builtin_amdgcn_s_wait_tensorcnt(2);   // current slab retired
            } else {
                __builtin_amdgcn_s_wait_tensorcnt(0);
            }
        }
        __syncthreads();
#else
        const __bf16* ga = A + (size_t)(tM + cpRow) * Kdim + i * KT + cpOff;
        const __bf16* gb = B + (size_t)(tN + cpRow) * Kdim + i * KT + cpOff;
        *(v16bf*)&lA[buf][cpRow * LDT + cpOff] = *(const v16bf*)ga;
        *(v16bf*)&lB[buf][cpRow * LDT + cpOff] = *(const v16bf*)gb;
        __syncthreads();
#endif

        {
            v16bf bf[2];
#pragma unroll
            for (int nt = 0; nt < 2; ++nt) {   // B frag: lane = col, K split by lane-half
                ((v8bf*)&bf[nt])[0] = *(const v8bf*)&lB[buf][(wn * 32 + nt * 16 + l16) * LDT + hl * 8];
                ((v8bf*)&bf[nt])[1] = *(const v8bf*)&lB[buf][(wn * 32 + nt * 16 + l16) * LDT + 16 + hl * 8];
            }
#pragma unroll
            for (int mt = 0; mt < 4; ++mt) {
                v16bf af;                     // A frag: lane = row, K {8h..}+{16+8h..}
                ((v8bf*)&af)[0] = *(const v8bf*)&lA[buf][(wm * 64 + mt * 16 + l16) * LDT + hl * 8];
                ((v8bf*)&af)[1] = *(const v8bf*)&lA[buf][(wm * 64 + mt * 16 + l16) * LDT + 16 + hl * 8];
#pragma unroll
                for (int nt = 0; nt < 2; ++nt)
                    acc[mt][nt] = __builtin_amdgcn_wmma_f32_16x16x32_bf16(
                        false, af, false, bf[nt], (short)0, acc[mt][nt], false, false);
            }
        }
        __syncthreads();
    }

#pragma unroll
    for (int mt = 0; mt < 4; ++mt)
#pragma unroll
        for (int nt = 0; nt < 2; ++nt)
#pragma unroll
            for (int r = 0; r < 8; ++r) {
                int row = tM + wm * 64 + mt * 16 + r + hl * 8;
                int col = tN + wn * 32 + nt * 16 + l16;
                float v = acc[mt][nt][r];
                if constexpr (MODE == 0) {
                    int b = row >> 11, s = row & (S_LEN - 1);
                    int h = col >> 6,  dh = col & (DH - 1);
                    ((__bf16*)C)[(((size_t)b * NH + h) * S_LEN + s) * DH + dh] = (__bf16)v;
                } else {
                    ((float*)C)[(size_t)row * Ndim + col] = v;
                }
            }
}

// ---------------------------------------------------------------- RoPE on Q,K
__global__ void k_rope(__bf16* __restrict__ q, __bf16* __restrict__ k, int total)
{
    int idx = blockIdx.x * blockDim.x + threadIdx.x;
    if (idx >= total) return;
    int i   = idx & 31;          // frequency index (DH/2 = 32)
    int bhs = idx >> 5;
    int s   = bhs & (S_LEN - 1); // layout [b,h,s,dh]
    float inv = __powf(10000.f, -(float)i * (1.f / 32.f));
    float sn, cs;
    __sincosf((float)s * inv, &sn, &cs);
    size_t base = (size_t)bhs * DH + 2 * i;
    float q1 = (float)q[base], q2 = (float)q[base + 1];
    q[base]     = (__bf16)(q1 * cs - q2 * sn);
    q[base + 1] = (__bf16)(q1 * sn + q2 * cs);
    float k1 = (float)k[base], k2 = (float)k[base + 1];
    k[base]     = (__bf16)(k1 * cs - k2 * sn);
    k[base + 1] = (__bf16)(k1 * sn + k2 * cs);
}

// ---------------------------------------------------------------- flash attention
// One wave = 16 query rows; 8 waves = 128 queries per workgroup.
// Per 32-key block: QK^T = 2x2 WMMAs, online softmax (16-lane shfl reductions),
// P via per-wave LDS (C-layout -> A-layout), V transposed into LDS, PV = 4 WMMAs.
__global__ __launch_bounds__(256) void k_flash(const __bf16* __restrict__ Q,
                                               const __bf16* __restrict__ K,
                                               const __bf16* __restrict__ V,
                                               __bf16* __restrict__ O)
{
    constexpr int PLD = 40;   // 16x32 P tile row stride (bf16)
    constexpr int VLD = 48;   // 64x32 V^T tile row stride (bf16, keeps 32B align)
    __shared__ alignas(32) __bf16 lP[8][16 * PLD];
    __shared__ alignas(32) __bf16 lV[8][64 * VLD];

    const int bh   = blockIdx.y;                  // b*NH + h
    const int q0   = blockIdx.x * 128;
    const int lane = threadIdx.x & 31, wave = threadIdx.x >> 5;
    const int hl = lane >> 4, l16 = lane & 15;
    const int qrow0 = q0 + wave * 16;

    const __bf16* Qb = Q + (size_t)bh * S_LEN * DH;
    const __bf16* Kb = K + (size_t)bh * S_LEN * DH;
    const __bf16* Vb = V + (size_t)bh * S_LEN * DH;
    __bf16* lPw = lP[wave];
    __bf16* lVw = lV[wave];

    // Q fragments (two K=32 chunks over dh), pre-scaled by 1/sqrt(DH)
    v16bf qf[2];
#pragma unroll
    for (int c = 0; c < 2; ++c) {
        const __bf16* qp = Qb + (size_t)(qrow0 + l16) * DH + c * 32;
        ((v8bf*)&qf[c])[0] = *(const v8bf*)(qp + hl * 8);
        ((v8bf*)&qf[c])[1] = *(const v8bf*)(qp + 16 + hl * 8);
#pragma unroll
        for (int e = 0; e < 16; ++e)
            qf[c][e] = (__bf16)((float)qf[c][e] * 0.125f);
    }

    const v8f vzero = {0.f, 0.f, 0.f, 0.f, 0.f, 0.f, 0.f, 0.f};
    float mrow[8], lrow[8];
    v8f o[4];
#pragma unroll
    for (int r = 0; r < 8; ++r) { mrow[r] = -3.0e38f; lrow[r] = 0.f; }
#pragma unroll
    for (int nt = 0; nt < 4; ++nt) o[nt] = vzero;

    for (int t0 = 0; t0 < qrow0 + 16; t0 += 32) {
        // scores: 16x32 tile = two 16x16 WMMA accs, each chaining K=0..63 in 2 steps
        float sv[2][8];
#pragma unroll
        for (int nt = 0; nt < 2; ++nt) {
            v8f s = vzero;
#pragma unroll
            for (int c = 0; c < 2; ++c) {
                v16bf kf;
                ((v8bf*)&kf)[0] = *(const v8bf*)(Kb + (size_t)(t0 + nt * 16 + l16) * DH + c * 32 + hl * 8);
                ((v8bf*)&kf)[1] = *(const v8bf*)(Kb + (size_t)(t0 + nt * 16 + l16) * DH + c * 32 + 16 + hl * 8);
                s = __builtin_amdgcn_wmma_f32_16x16x32_bf16(false, qf[c], false, kf,
                                                            (short)0, s, false, false);
            }
            int t = t0 + nt * 16 + l16;          // this lane's key column
#pragma unroll
            for (int r = 0; r < 8; ++r) {
                int qr = qrow0 + r + hl * 8;     // this VGPR's query row
                sv[nt][r] = (t <= qr) ? s[r] : -1.0e30f;   // causal mask
            }
        }
        // online softmax state update
        float newm[8], alpha[8], psum[8];
#pragma unroll
        for (int r = 0; r < 8; ++r) {
            float m = fmaxf(sv[0][r], sv[1][r]);
#pragma unroll
            for (int off = 1; off < 16; off <<= 1)
                m = fmaxf(m, __shfl_xor(m, off, 32));    // row lives in one 16-lane half
            newm[r]  = fmaxf(mrow[r], m);
            alpha[r] = __expf(mrow[r] - newm[r]);
            mrow[r]  = newm[r];
            psum[r]  = 0.f;
        }
#pragma unroll
        for (int nt = 0; nt < 2; ++nt)
#pragma unroll
            for (int r = 0; r < 8; ++r) {
                float p = __expf(sv[nt][r] - newm[r]);
                psum[r] += p;
                lPw[(r + hl * 8) * PLD + nt * 16 + l16] = (__bf16)p;  // C -> A relayout
            }
#pragma unroll
        for (int r = 0; r < 8; ++r) {
            float ps = psum[r];
#pragma unroll
            for (int off = 1; off < 16; off <<= 1)
                ps += __shfl_xor(ps, off, 32);
            lrow[r] = lrow[r] * alpha[r] + ps;
        }
#pragma unroll
        for (int nt = 0; nt < 4; ++nt)
#pragma unroll
            for (int r = 0; r < 8; ++r)
                o[nt][r] *= alpha[r];

        // stage V^T into LDS: lV[d][k], lane = key
        {
            const __bf16* vr = Vb + (size_t)(t0 + lane) * DH;
            v16bf v0 = *(const v16bf*)(vr);
            v16bf v1 = *(const v16bf*)(vr + 16);
            v16bf v2 = *(const v16bf*)(vr + 32);
            v16bf v3 = *(const v16bf*)(vr + 48);
#pragma unroll
            for (int d = 0; d < 16; ++d) {
                lVw[(d)      * VLD + lane] = v0[d];
                lVw[(d + 16) * VLD + lane] = v1[d];
                lVw[(d + 32) * VLD + lane] = v2[d];
                lVw[(d + 48) * VLD + lane] = v3[d];
            }
        }

        // O += P * V  (P: 16x32 A-frag from LDS, V^T: B-frags, 4 n-tiles over dh)
        v16bf pf;
        ((v8bf*)&pf)[0] = *(const v8bf*)&lPw[l16 * PLD + hl * 8];
        ((v8bf*)&pf)[1] = *(const v8bf*)&lPw[l16 * PLD + 16 + hl * 8];
#pragma unroll
        for (int nt = 0; nt < 4; ++nt) {
            v16bf vf;
            ((v8bf*)&vf)[0] = *(const v8bf*)&lVw[(nt * 16 + l16) * VLD + hl * 8];
            ((v8bf*)&vf)[1] = *(const v8bf*)&lVw[(nt * 16 + l16) * VLD + 16 + hl * 8];
            o[nt] = __builtin_amdgcn_wmma_f32_16x16x32_bf16(false, pf, false, vf,
                                                            (short)0, o[nt], false, false);
        }
    }

    // epilogue: normalize and store to [b, s, h*DH + d] for the wo GEMM
    const int b = bh >> 4, h = bh & (NH - 1);
#pragma unroll
    for (int r = 0; r < 8; ++r) {
        float inv = 1.f / lrow[r];
        int srow = qrow0 + r + hl * 8;
        size_t base = ((size_t)b * S_LEN + srow) * DMODEL + h * DH;
#pragma unroll
        for (int nt = 0; nt < 4; ++nt)
            O[base + nt * 16 + l16] = (__bf16)(o[nt][r] * inv);
    }
}

// ---------------------------------------------------------------- launch
extern "C" void kernel_launch(void* const* d_in, const int* in_sizes, int n_in,
                              void* d_out, int out_size, void* d_ws, size_t ws_size,
                              hipStream_t stream)
{
    const float* x  = (const float*)d_in[0];
    const float* wq = (const float*)d_in[1];
    const float* wk = (const float*)d_in[2];
    const float* wv = (const float*)d_in[3];
    const float* wo = (const float*)d_in[4];

    const size_t nx = (size_t)ROWS * DMODEL;      // 8M activations
    const size_t nw = (size_t)DMODEL * DMODEL;    // 1M per weight

    __bf16* xb  = (__bf16*)d_ws;          // x (bf16), row-major [r, d]
    __bf16* wqb = xb  + nx;
    __bf16* wkb = wqb + nw;
    __bf16* wvb = wkb + nw;
    __bf16* wob = wvb + nw;
    __bf16* qb  = wob + nw;               // [b,h,s,dh]
    __bf16* kb  = qb  + nx;               // [b,h,s,dh]
    __bf16* vb  = kb  + nx;               // [b,h,s,dh]
    __bf16* ab  = vb  + nx;               // attn out, [r, d]
    // total: 44M bf16 = 88 MB of workspace

    k_cvt_bf16<<<dim3((unsigned)((nx / 4 + 255) / 256)), 256, 0, stream>>>(x,  xb,  (int)nx);
    k_cvt_bf16<<<dim3((unsigned)((nw / 4 + 255) / 256)), 256, 0, stream>>>(wq, wqb, (int)nw);
    k_cvt_bf16<<<dim3((unsigned)((nw / 4 + 255) / 256)), 256, 0, stream>>>(wk, wkb, (int)nw);
    k_cvt_bf16<<<dim3((unsigned)((nw / 4 + 255) / 256)), 256, 0, stream>>>(wv, wvb, (int)nw);
    k_cvt_bf16<<<dim3((unsigned)((nw / 4 + 255) / 256)), 256, 0, stream>>>(wo, wob, (int)nw);

    dim3 gg(DMODEL / 128, ROWS / 128);    // (8, 64) workgroups of 256 threads
    k_gemm<0><<<gg, 256, 0, stream>>>(xb, wqb, (void*)qb, DMODEL, DMODEL, ROWS);
    k_gemm<0><<<gg, 256, 0, stream>>>(xb, wkb, (void*)kb, DMODEL, DMODEL, ROWS);
    k_gemm<0><<<gg, 256, 0, stream>>>(xb, wvb, (void*)vb, DMODEL, DMODEL, ROWS);

    int rtot = BATCH * NH * S_LEN * 32;   // one thread per (b,h,s,freq)
    k_rope<<<dim3((rtot + 255) / 256), 256, 0, stream>>>(qb, kb, rtot);

    k_flash<<<dim3(S_LEN / 128, BATCH * NH), 256, 0, stream>>>(qb, kb, vb, ab);

    k_gemm<1><<<gg, 256, 0, stream>>>(ab, wob, d_out, DMODEL, DMODEL, ROWS);
}
